// PredNoiseAndRoutLUTsWR_62302795595978
// MI455X (gfx1250) — matrix operand order repaired and checked by reading, not compiled, not run on hardware
//
#include <hip/hip_runtime.h>

// ---------------------------------------------------------------------------
// Types for CDNA5 WMMA (wave32). 16x16x32 bf16 -> f32 accumulate.
// ---------------------------------------------------------------------------
typedef __attribute__((ext_vector_type(16))) __bf16 v16bf;
typedef __attribute__((ext_vector_type(8)))  float  v8f;
typedef __attribute__((ext_vector_type(4)))  int    v4i;

// address-space-qualified vector pointees for the async-DMA builtin
typedef __attribute__((address_space(1))) v4i as1_v4i;   // global
typedef __attribute__((address_space(3))) v4i as3_v4i;   // LDS

__device__ __forceinline__ __bf16 f2bf(float x) { return (__bf16)x; }

// ---------------------------------------------------------------------------
// Weight staging into LDS (f32) via CDNA5 async global->LDS DMA
// (GLOBAL_LOAD_ASYNC_TO_LDS_B128, ASYNCcnt). Caller must __syncthreads().
// ---------------------------------------------------------------------------
__device__ __forceinline__ void stage_weights(const float* __restrict__ w,
                                              float* lds_wf, int nW,
                                              int tid, int nthreads)
{
#if __has_builtin(__builtin_amdgcn_global_load_async_to_lds_b128)
    const int nVec = nW >> 2;                 // 16-byte packets
    for (int i = tid; i < nVec; i += nthreads) {
        __builtin_amdgcn_global_load_async_to_lds_b128(
            (as1_v4i*)(w + (size_t)i * 4),
            (as3_v4i*)(lds_wf + (size_t)i * 4),
            0, 0);
    }
    for (int i = (nVec << 2) + tid; i < nW; i += nthreads)   // tail (<4 floats)
        lds_wf[i] = w[i];
#if __has_builtin(__builtin_amdgcn_s_wait_asynccnt)
    __builtin_amdgcn_s_wait_asynccnt(0);
#else
    asm volatile("s_wait_asynccnt 0" ::: "memory");
#endif
#else
    for (int i = tid; i < nW; i += nthreads)
        lds_wf[i] = w[i];
#endif
}

// ---------------------------------------------------------------------------
// Common wave-uniform tile decomposition.
//   M-tile = 16 pixels along x (W % 16 == 0, so a tile never straddles a row).
// ---------------------------------------------------------------------------
struct TileInfo { int bImg, y, x0; bool valid; };

__device__ __forceinline__ TileInfo tile_decode(int tile, int Bn, int H, int W)
{
    TileInfo t;
    const int HW = H * W;
    const int nTiles = (Bn * HW) >> 4;
    t.valid = (tile < nTiles);
    const int p0 = tile << 4;
    t.bImg = p0 / HW;
    const int rr = p0 - t.bImg * HW;
    t.y  = rr / W;
    t.x0 = rr - t.y * W;
    return t;
}

// per-lane A-fragment K-pattern: element h -> K offset within a 32-chunk
__device__ __forceinline__ int a_koff(int h, int half)
{
    return ((h < 8) ? 0 : 16) + half * 8 + (h & 7);
}

// ---------------------------------------------------------------------------
// Tap-split 3x3 conv (pad 1) for Cin % 32 == 0 (conv_out: 320, f2: 64).
// 9 tap-GEMMs with K = channels: no divisions in the gather; the bounds
// predicate is per-tap/per-lane (one cndmask per element, no branches).
// All gather offsets are int32 (fits: Cin*HW < 2^31) to avoid 64-bit VALU.
// B index: w[n][c][tap] -> lds_wf[n*Cin*9 + c*9 + tap].
// ---------------------------------------------------------------------------
template <bool RELU>
__global__ void __launch_bounds__(256)
wmma_conv3_tap_kernel(const float* __restrict__ in, const float* __restrict__ w,
                      const float* __restrict__ bias, float* __restrict__ out,
                      int Bn, int Cin, int Cout, int H, int W)
{
    extern __shared__ float lds_wf[];                  // [Cout * Cin * 9] f32
    const int Ktot = Cin * 9;
    const int tid  = threadIdx.x;

    stage_weights(w, lds_wf, Cout * Ktot, tid, blockDim.x);
    __syncthreads();

    const int lane = tid & 31;
    const int tile = blockIdx.x * (blockDim.x >> 5) + (tid >> 5);
    const TileInfo T = tile_decode(tile, Bn, H, W);
    if (!T.valid) return;                              // wave-uniform

    const int HW   = H * W;
    const int half = lane >> 4;
    const int m    = lane & 15;
    const int nG   = blockIdx.y * 16 + (lane & 15);
    const bool nOK = (nG < Cout);
    const int x    = T.x0 + m;

    const float* __restrict__ inB = in + (size_t)T.bImg * Cin * HW;
    const float* __restrict__ wRow = lds_wf + (nOK ? nG : 0) * Ktot;

    // per-lane invariant A channel-offsets (int32), computed once
    int aOff[16];
#pragma unroll
    for (int h = 0; h < 16; ++h) aOff[h] = a_koff(h, half) * HW;

    const int cChunks = Cin >> 5;                      // Cin % 32 == 0
    v8f acc = {0.f, 0.f, 0.f, 0.f, 0.f, 0.f, 0.f, 0.f};

#pragma unroll
    for (int tap = 0; tap < 9; ++tap) {
        const int dy = tap / 3 - 1;                    // compile-time per unrolled tap
        const int dx = tap % 3 - 1;
        const int yy = T.y + dy;
        const int xx = x + dx;
        const bool ok = ((unsigned)yy < (unsigned)H) & ((unsigned)xx < (unsigned)W);
        const int rowoff = ok ? (yy * W + xx) : 0;     // clamped, always loadable

        for (int cc = 0; cc < cChunks; ++cc) {
            const int cb = (cc << 5) * HW + rowoff;    // int32 chunk base

            // ---- A fragment: 16 channel-strided loads, one cndmask each ----
            v16bf a;
#pragma unroll
            for (int h = 0; h < 16; ++h) {
                const float v = inB[cb + aOff[h]];
                a[h] = f2bf(ok ? v : 0.f);
            }

            // ---- B fragment: LDS reads, stride 9 floats, immediate offsets ----
            v16bf bm;
            const float* wp = wRow + (cc << 5) * 9 + tap;
#pragma unroll
            for (int h = 0; h < 16; ++h) {
                const float v = wp[a_koff(h, half) * 9];
                bm[h] = f2bf(nOK ? v : 0.f);
            }

            acc = __builtin_amdgcn_wmma_f32_16x16x32_bf16(
                false, a, false, bm, (short)0, acc, false, false);
        }
    }

    if (nOK) {
        const float bv = bias[nG];
        float* __restrict__ op =
            out + ((size_t)T.bImg * Cout + nG) * HW + T.y * W + T.x0 + half * 8;
#pragma unroll
        for (int i = 0; i < 8; ++i) {
            float v = acc[i] + bv;
            if (RELU) v = fmaxf(v, 0.f);
            op[i] = v;
        }
    }
}

// ---------------------------------------------------------------------------
// Generic packed-K implicit-GEMM conv (small Cin: router 3x3, f1, and 1x1).
// Clamped unconditional loads + selects (cndmask, no branches); int32 offsets.
// ---------------------------------------------------------------------------
template <int KS, int PAD, bool RELU>
__global__ void __launch_bounds__(256)
wmma_conv_kernel(const float* __restrict__ in, const float* __restrict__ w,
                 const float* __restrict__ bias, float* __restrict__ out,
                 int Bn, int Cin, int Cout, int H, int W)
{
    constexpr int KTAPS = KS * KS;
    extern __shared__ float lds_wf[];                  // [Cout * Ktot] f32
    const int Ktot = Cin * KTAPS;
    const int tid  = threadIdx.x;

    stage_weights(w, lds_wf, Cout * Ktot, tid, blockDim.x);
    __syncthreads();

    const int lane = tid & 31;
    const int tile = blockIdx.x * (blockDim.x >> 5) + (tid >> 5);
    const TileInfo T = tile_decode(tile, Bn, H, W);
    if (!T.valid) return;                              // wave-uniform

    const int HW   = H * W;
    const int half = lane >> 4;
    const int m    = lane & 15;
    const int nG   = blockIdx.y * 16 + (lane & 15);
    const bool nOK = (nG < Cout);
    const int x    = T.x0 + m;
    const int y    = T.y;

    const float* __restrict__ inB = in + (size_t)T.bImg * Cin * HW;
    const float* __restrict__ wRow = lds_wf + (nOK ? nG : 0) * Ktot;
    const int baseYX  = y * W + x;
    const int nChunks = (Ktot + 31) >> 5;

    v8f acc = {0.f, 0.f, 0.f, 0.f, 0.f, 0.f, 0.f, 0.f};

    for (int kc = 0; kc < nChunks; ++kc) {
        const int kBase = kc << 5;

        // ---- A fragment: clamped loads + selects (no branches) ----
        v16bf a;
#pragma unroll
        for (int h = 0; h < 16; ++h) {
            const int k   = kBase + a_koff(h, half);
            const int c   = k / KTAPS;                 // constant divisor (1 for 1x1)
            const int tap = k - c * KTAPS;
            bool ok = (k < Ktot);
            int off;
            if (KS == 1) {
                off = c * HW + baseYX;
            } else {
                const int ty = tap / KS;
                const int dy = ty - PAD;
                const int dx = (tap - ty * KS) - PAD;
                const int yy = y + dy;
                const int xx = x + dx;
                ok = ok & ((unsigned)yy < (unsigned)H) & ((unsigned)xx < (unsigned)W);
                off = c * HW + yy * W + xx;
            }
            off = ok ? off : 0;                        // clamp; always loadable
            const float v = inB[off];
            a[h] = f2bf(ok ? v : 0.f);
        }

        // ---- B fragment from LDS: clamped + select ----
        v16bf bm;
#pragma unroll
        for (int h = 0; h < 16; ++h) {
            const int k = kBase + a_koff(h, half);
            const bool ok = (k < Ktot) & nOK;
            const float v = wRow[ok ? k : 0];
            bm[h] = f2bf(ok ? v : 0.f);
        }

        acc = __builtin_amdgcn_wmma_f32_16x16x32_bf16(
            false, a, false, bm, (short)0, acc, false, false);
    }

    if (nOK) {
        const float bv = bias[nG];
        float* __restrict__ op =
            out + ((size_t)T.bImg * Cout + nG) * HW + y * W + T.x0 + half * 8;
#pragma unroll
        for (int i = 0; i < 8; ++i) {
            float v = acc[i] + bv;
            if (RELU) v = fmaxf(v, 0.f);
            op[i] = v;
        }
    }
}

// ---------------------------------------------------------------------------
// ConvTranspose2d, kernel 4x4, stride 2, pad 1 (C = 9): direct VALU.
// ---------------------------------------------------------------------------
__global__ void convt_kernel(const float* __restrict__ in, const float* __restrict__ w,
                             const float* __restrict__ bias, float* __restrict__ out,
                             int Bn, int C, int Hin, int Win)
{
    const int Ho = Hin * 2, Wo = Win * 2;
    const size_t total = (size_t)Bn * C * Ho * Wo;
    size_t idx = (size_t)blockIdx.x * blockDim.x + threadIdx.x;
    if (idx >= total) return;

    const int xo = (int)(idx % Wo); size_t t = idx / Wo;
    const int yo = (int)(t % Ho);   t /= Ho;
    const int co = (int)(t % C);
    const int b  = (int)(t / C);

    float accv = bias[co];
#pragma unroll
    for (int ky = 0; ky < 4; ++ky) {
        const int ny = yo + 1 - ky;
        if (ny & 1) continue;
        const int iy = ny >> 1;
        if (iy < 0 || iy >= Hin) continue;
#pragma unroll
        for (int kx = 0; kx < 4; ++kx) {
            const int nx = xo + 1 - kx;
            if (nx & 1) continue;
            const int ix = nx >> 1;
            if (ix < 0 || ix >= Win) continue;
            for (int ci = 0; ci < C; ++ci) {
                accv += in[(((size_t)b * C + ci) * Hin + iy) * Win + ix]
                      * w[(((size_t)ci * C + co) * 4 + ky) * 4 + kx];
            }
        }
    }
    out[idx] = accv;
}

// ---------------------------------------------------------------------------
// Softmax over K per pixel.
// ---------------------------------------------------------------------------
__global__ void softmax_k_kernel(const float* __restrict__ in, float* __restrict__ out,
                                 int Bn, int K, int HW)
{
    size_t idx = (size_t)blockIdx.x * blockDim.x + threadIdx.x;
    const size_t total = (size_t)Bn * HW;
    if (idx >= total) return;
    const int b  = (int)(idx / HW);
    const int hw = (int)(idx - (size_t)b * HW);
    const float* p = in + (size_t)b * K * HW + hw;
    float mx = -3.4e38f;
    for (int k = 0; k < K; ++k) mx = fmaxf(mx, p[(size_t)k * HW]);
    float s = 0.f;
    for (int k = 0; k < K; ++k) s += __expf(p[(size_t)k * HW] - mx);
    const float inv = 1.f / s;
    float* q = out + (size_t)b * K * HW + hw;
    for (int k = 0; k < K; ++k) q[(size_t)k * HW] = __expf(p[(size_t)k * HW] - mx) * inv;
}

// ---------------------------------------------------------------------------
// R reductions: sum(R^2), sum|dh| (along W), sum|dw| (along H).
// LDS float atomics (ds_add_f32) + one global atomic per block.
// ---------------------------------------------------------------------------
__global__ void r_reduce_kernel(const float* __restrict__ R, float* __restrict__ accum,
                                int Bn, int K, int H, int W)
{
    __shared__ float sacc[3];
    if (threadIdx.x < 3) sacc[threadIdx.x] = 0.f;
    __syncthreads();

    const size_t total = (size_t)Bn * K * H * W;
    float s0 = 0.f, s1 = 0.f, s2 = 0.f;
    for (size_t i = (size_t)blockIdx.x * blockDim.x + threadIdx.x; i < total;
         i += (size_t)gridDim.x * blockDim.x) {
        const float v = R[i];
        s0 += v * v;
        const int x = (int)(i % W);
        const int y = (int)((i / W) % H);
        if (x < W - 1) s1 += fabsf(v - R[i + 1]);
        if (y < H - 1) s2 += fabsf(v - R[i + W]);
    }
    atomicAdd(&sacc[0], s0);
    atomicAdd(&sacc[1], s1);
    atomicAdd(&sacc[2], s2);
    __syncthreads();
    if (threadIdx.x < 3) atomicAdd(&accum[threadIdx.x], sacc[threadIdx.x]);
}

// ---------------------------------------------------------------------------
// 3D-LUT TV / monotonicity reduction over luts (K,3,D,D,D).
// ---------------------------------------------------------------------------
__global__ void lut_reduce_kernel(const float* __restrict__ luts, float* __restrict__ accum,
                                  int K, int D)
{
    __shared__ float sacc[6];
    if (threadIdx.x < 6) sacc[threadIdx.x] = 0.f;
    __syncthreads();

    const size_t total = (size_t)K * 3 * D * D * D;
    float tr = 0.f, tg = 0.f, tb = 0.f, mr = 0.f, mg = 0.f, mb = 0.f;
    for (size_t i = (size_t)blockIdx.x * blockDim.x + threadIdx.x; i < total;
         i += (size_t)gridDim.x * blockDim.x) {
        const int ir = (int)(i % D);
        const int ig = (int)((i / D) % D);
        const int ib = (int)((i / ((size_t)D * D)) % D);
        const float v = luts[i];
        if (ir < D - 1) {
            const float d = v - luts[i + 1];
            const float wgt = (ir == 0 || ir == D - 2) ? 2.f : 1.f;
            tr += d * d * wgt;  mr += fmaxf(d, 0.f);
        }
        if (ig < D - 1) {
            const float d = v - luts[i + D];
            const float wgt = (ig == 0 || ig == D - 2) ? 2.f : 1.f;
            tg += d * d * wgt;  mg += fmaxf(d, 0.f);
        }
        if (ib < D - 1) {
            const float d = v - luts[i + (size_t)D * D];
            const float wgt = (ib == 0 || ib == D - 2) ? 2.f : 1.f;
            tb += d * d * wgt;  mb += fmaxf(d, 0.f);
        }
    }
    atomicAdd(&sacc[0], tr); atomicAdd(&sacc[1], tg); atomicAdd(&sacc[2], tb);
    atomicAdd(&sacc[3], mr); atomicAdd(&sacc[4], mg); atomicAdd(&sacc[5], mb);
    __syncthreads();
    if (threadIdx.x < 6) atomicAdd(&accum[3 + threadIdx.x], sacc[threadIdx.x]);
}

// ---------------------------------------------------------------------------
// Fused trilinear LUT sampling + einsum('kcbhw,bkhw->bchw').
// ---------------------------------------------------------------------------
__global__ void lut_apply_kernel(const float* __restrict__ img, const float* __restrict__ luts,
                                 const float* __restrict__ R, float* __restrict__ out,
                                 int Bn, int K, int D, int H, int W)
{
    size_t idx = (size_t)blockIdx.x * blockDim.x + threadIdx.x;
    const size_t total = (size_t)Bn * H * W;
    if (idx >= total) return;
    const int HW = H * W;
    const int b  = (int)(idx / HW);
    const int hw = (int)(idx - (size_t)b * HW);

    float pos[3], fr[3]; int i0[3], i1[3];
#pragma unroll
    for (int c = 0; c < 3; ++c) {
        const float v01 = img[((size_t)b * 3 + c) * HW + hw] * 0.5f + 0.5f;
        pos[c] = v01 * (float)(D - 1);
        const float fl = floorf(pos[c]);
        int ii = (int)fl;
        ii = ii < 0 ? 0 : (ii > D - 1 ? D - 1 : ii);
        i0[c] = ii;
        int jj = ii + 1;
        i1[c] = jj > D - 1 ? D - 1 : jj;
        fr[c] = pos[c] - fl;
    }
    const float frx = fr[0], fgx = fr[1], fbx = fr[2];
    const float w000 = (1 - fbx) * (1 - fgx) * (1 - frx);
    const float w001 = (1 - fbx) * (1 - fgx) * frx;
    const float w010 = (1 - fbx) * fgx * (1 - frx);
    const float w011 = (1 - fbx) * fgx * frx;
    const float w100 = fbx * (1 - fgx) * (1 - frx);
    const float w101 = fbx * (1 - fgx) * frx;
    const float w110 = fbx * fgx * (1 - frx);
    const float w111 = fbx * fgx * frx;

    const int r0 = i0[0], r1 = i1[0], g0 = i0[1], g1 = i1[1], b0 = i0[2], b1 = i1[2];
    float acc0 = 0.f, acc1 = 0.f, acc2 = 0.f;

    for (int k = 0; k < K; ++k) {
        const float wk = R[((size_t)b * K + k) * HW + hw];
#pragma unroll
        for (int c = 0; c < 3; ++c) {
            const float* L = luts + ((size_t)k * 3 + c) * D * D * D;
            const size_t sB0 = (size_t)b0 * D * D, sB1 = (size_t)b1 * D * D;
            const size_t sG0 = (size_t)g0 * D,     sG1 = (size_t)g1 * D;
            const float val =
                L[sB0 + sG0 + r0] * w000 + L[sB0 + sG0 + r1] * w001 +
                L[sB0 + sG1 + r0] * w010 + L[sB0 + sG1 + r1] * w011 +
                L[sB1 + sG0 + r0] * w100 + L[sB1 + sG0 + r1] * w101 +
                L[sB1 + sG1 + r0] * w110 + L[sB1 + sG1 + r1] * w111;
            const float contrib = wk * val;
            if (c == 0) acc0 += contrib; else if (c == 1) acc1 += contrib; else acc2 += contrib;
        }
    }
    out[((size_t)b * 3 + 0) * HW + hw] = acc0;
    out[((size_t)b * 3 + 1) * HW + hw] = acc1;
    out[((size_t)b * 3 + 2) * HW + hw] = acc2;
}

// ---------------------------------------------------------------------------
// Small helpers: zero accumulators, assemble reg_loss scalar.
// ---------------------------------------------------------------------------
__global__ void init_accum_kernel(float* accum)
{
    if (threadIdx.x < 16) accum[threadIdx.x] = 0.f;
}

__global__ void finalize_kernel(const float* __restrict__ accum, float* __restrict__ out_reg,
                                int Bn, int K, int H, int W, int D)
{
    if (threadIdx.x == 0 && blockIdx.x == 0) {
        const float NR   = (float)Bn * K * H * W;
        const float wn   = accum[0] / NR;
        const float tvR  = accum[1] / ((float)Bn * K * H * (W - 1))
                         + accum[2] / ((float)Bn * K * (H - 1) * W);
        const float den  = 3.f * D * D * (D - 1);
        const float tvc  = (accum[3] + accum[4] + accum[5]) / den;
        const float mnc  = (accum[6] + accum[7] + accum[8]) / den;
        *out_reg = 0.001f * (wn + tvc + tvR) + 10.f * mnc;
    }
}

// ---------------------------------------------------------------------------
// Host side
// ---------------------------------------------------------------------------
static void launch_conv(const float* in, const float* w, const float* b, float* out,
                        int Bn, int Cin, int Cout, int H, int W,
                        int KS, int relu, hipStream_t s)
{
    const int Ktot   = Cin * KS * KS;
    const int nTiles = (Bn * H * W + 15) / 16;
    dim3 grid((nTiles + 7) / 8, (Cout + 15) / 16);
    const size_t ldsB = (size_t)Cout * Ktot * sizeof(float);
    if (KS == 3 && (Cin & 31) == 0) {
        // tap-split path: division-free gather, per-tap bounds predicate
        if (relu)
            wmma_conv3_tap_kernel<true ><<<grid, 256, ldsB, s>>>(in, w, b, out, Bn, Cin, Cout, H, W);
        else
            wmma_conv3_tap_kernel<false><<<grid, 256, ldsB, s>>>(in, w, b, out, Bn, Cin, Cout, H, W);
    } else if (KS == 3) {
        if (relu)
            wmma_conv_kernel<3, 1, true ><<<grid, 256, ldsB, s>>>(in, w, b, out, Bn, Cin, Cout, H, W);
        else
            wmma_conv_kernel<3, 1, false><<<grid, 256, ldsB, s>>>(in, w, b, out, Bn, Cin, Cout, H, W);
    } else {
        if (relu)
            wmma_conv_kernel<1, 0, true ><<<grid, 256, ldsB, s>>>(in, w, b, out, Bn, Cin, Cout, H, W);
        else
            wmma_conv_kernel<1, 0, false><<<grid, 256, ldsB, s>>>(in, w, b, out, Bn, Cin, Cout, H, W);
    }
}

extern "C" void kernel_launch(void* const* d_in, const int* in_sizes, int n_in,
                              void* d_out, int out_size, void* d_ws, size_t ws_size,
                              hipStream_t stream)
{
    (void)in_sizes; (void)n_in; (void)out_size; (void)ws_size;

    const float* feat       = (const float*)d_in[0];
    const float* img        = (const float*)d_in[1];
    const float* conv_out_w = (const float*)d_in[2];
    const float* conv_out_b = (const float*)d_in[3];
    const float* r1_w = (const float*)d_in[4];  const float* r1_b = (const float*)d_in[5];
    const float* r2_w = (const float*)d_in[6];  const float* r2_b = (const float*)d_in[7];
    const float* rt1_w = (const float*)d_in[8]; const float* rt1_b = (const float*)d_in[9];
    const float* r3_w = (const float*)d_in[10]; const float* r3_b = (const float*)d_in[11];
    const float* rt2_w = (const float*)d_in[12];const float* rt2_b = (const float*)d_in[13];
    const float* r4_w = (const float*)d_in[14]; const float* r4_b = (const float*)d_in[15];
    const float* rt3_w = (const float*)d_in[16];const float* rt3_b = (const float*)d_in[17];
    const float* luts = (const float*)d_in[18];
    const float* f1_w = (const float*)d_in[19]; const float* f1_b = (const float*)d_in[20];
    const float* f2_w = (const float*)d_in[21]; const float* f2_b = (const float*)d_in[22];

    const int Bn = 4, K = 9, D = 33;
    const int Hf = 64, Wf = 64;      // feature-map res
    const int Hi = 512, Wi = 512;    // image res
    const int CF = 320;

    // workspace layout (floats)
    float* ws     = (float*)d_ws;
    const size_t nR = (size_t)Bn * K * Hi * Wi;          // 9,437,184
    float* buf0   = ws;
    float* buf1   = buf0 + nR;
    float* lutpre = buf1 + nR;                           // (B,3,512,512)
    float* hbuf   = lutpre + (size_t)Bn * 3 * Hi * Wi;   // (B,64,512,512)
    float* accum  = hbuf + (size_t)Bn * 64 * Hi * Wi;    // 16 floats

    // output layout
    float* out_noise = (float*)d_out;                                   // (4,4,64,64)
    float* out_lut   = out_noise + (size_t)Bn * 4 * Hf * Wf;            // (4,3,512,512)
    float* out_reg   = out_lut + (size_t)Bn * 3 * Hi * Wi;              // scalar

    init_accum_kernel<<<1, 32, 0, stream>>>(accum);

    // noise_pred = conv2d(feat, conv_out_w, b, pad=1)   [tap-split WMMA, K=2880]
    launch_conv(feat, conv_out_w, conv_out_b, out_noise, Bn, CF, 4, Hf, Wf, 3, 0, stream);

    // router chain
    launch_conv(feat, r1_w, r1_b, buf0, Bn, CF, K, Hf, Wf, 1, 1, stream);   // relu(1x1 320->9)
    launch_conv(buf0, r2_w, r2_b, buf1, Bn, K, K, Hf, Wf, 3, 1, stream);    // relu(3x3 9->9)

    {   // rt1: 64 -> 128
        const size_t tot = (size_t)Bn * K * 128 * 128;
        convt_kernel<<<(unsigned)((tot + 255) / 256), 256, 0, stream>>>(buf1, rt1_w, rt1_b, buf0, Bn, K, Hf, Wf);
    }
    launch_conv(buf0, r3_w, r3_b, buf1, Bn, K, K, 128, 128, 3, 1, stream);
    {   // rt2: 128 -> 256
        const size_t tot = (size_t)Bn * K * 256 * 256;
        convt_kernel<<<(unsigned)((tot + 255) / 256), 256, 0, stream>>>(buf1, rt2_w, rt2_b, buf0, Bn, K, 128, 128);
    }
    launch_conv(buf0, r4_w, r4_b, buf1, Bn, K, K, 256, 256, 3, 1, stream);
    {   // rt3: 256 -> 512
        const size_t tot = (size_t)Bn * K * 512 * 512;
        convt_kernel<<<(unsigned)((tot + 255) / 256), 256, 0, stream>>>(buf1, rt3_w, rt3_b, buf0, Bn, K, 256, 256);
    }

    // R = softmax over K; R-based reductions
    {
        const size_t tot = (size_t)Bn * Hi * Wi;
        softmax_k_kernel<<<(unsigned)((tot + 255) / 256), 256, 0, stream>>>(buf0, buf1, Bn, K, Hi * Wi);
    }
    r_reduce_kernel<<<4096, 256, 0, stream>>>(buf1, accum, Bn, K, Hi, Wi);

    // trilinear LUT sample + blend by R
    {
        const size_t tot = (size_t)Bn * Hi * Wi;
        lut_apply_kernel<<<(unsigned)((tot + 255) / 256), 256, 0, stream>>>(img, luts, buf1, lutpre, Bn, K, D, Hi, Wi);
    }

    // refinement convs (dominant FLOPs) — WMMA
    launch_conv(lutpre, f1_w, f1_b, hbuf, Bn, 3, 64, Hi, Wi, 3, 1, stream);     // packed-K (Cin=3)
    launch_conv(hbuf, f2_w, f2_b, out_lut, Bn, 64, 3, Hi, Wi, 3, 0, stream);    // tap-split (Cin=64)

    // LUT regularizers + final scalar
    lut_reduce_kernel<<<1024, 256, 0, stream>>>(luts, accum, K, D);
    finalize_kernel<<<1, 32, 0, stream>>>(accum, out_reg, Bn, K, Hi, Wi, D);
}
